// MerlotReserve_36490042147498
// MI455X (gfx1250) — matrix operand order repaired and checked by reading, not compiled, not run on hardware
//
#include <hip/hip_runtime.h>

// MerlotReserve 2-layer transformer forward for gfx1250 (MI455X).
// All matmuls via v_wmma_f32_16x16x32_f16 (wave32 WMMA), flash-attention with
// online softmax, f32 residual stream. Compute-bound (~241 GFLOP, <1GB HBM).

#define HDIM 1024
#define SEQ  1024
#define BATCH 4
#define NHEAD 16
#define DPH  64
#define EXPH 4096
#define LAYERS 2

typedef __attribute__((ext_vector_type(16))) _Float16 v16h;
typedef __attribute__((ext_vector_type(8)))  _Float16 v8h;
typedef __attribute__((ext_vector_type(2)))  _Float16 v2h;
typedef __attribute__((ext_vector_type(8)))  float    v8f;

static __device__ inline v16h combine8(v8h lo, v8h hi) {
    return __builtin_shufflevector(lo, hi, 0,1,2,3,4,5,6,7,8,9,10,11,12,13,14,15);
}

// ---------------------------------------------------------------- f32 -> f16
__global__ void cvt_f32_f16(const float* __restrict__ src, _Float16* __restrict__ dst, int n) {
    int i = blockIdx.x * blockDim.x + threadIdx.x;
    if (i < n) dst[i] = (_Float16)src[i];
}

// ---------------------------------------------------------------- LayerNorm
// one block per row (H=1024), 256 threads x 4 floats, f16 output
__global__ __launch_bounds__(256) void layernorm_f16k(
    const float* __restrict__ x, const float* __restrict__ g,
    const float* __restrict__ b, _Float16* __restrict__ out)
{
    __shared__ float ssum[8], ssq[8];
    const int row = blockIdx.x;
    const int tid = threadIdx.x;
    const float4 xv = *(const float4*)(x + (size_t)row * HDIM + tid * 4);
    float s = xv.x + xv.y + xv.z + xv.w;
    float q = xv.x*xv.x + xv.y*xv.y + xv.z*xv.z + xv.w*xv.w;
    for (int m = 16; m >= 1; m >>= 1) {
        s += __shfl_xor(s, m, 32);
        q += __shfl_xor(q, m, 32);
    }
    if ((tid & 31) == 0) { ssum[tid >> 5] = s; ssq[tid >> 5] = q; }
    __syncthreads();
    float ts = 0.f, tq = 0.f;
    #pragma unroll
    for (int i = 0; i < 8; i++) { ts += ssum[i]; tq += ssq[i]; }
    const float mean = ts * (1.0f / HDIM);
    const float var  = tq * (1.0f / HDIM) - mean * mean;
    const float inv  = rsqrtf(var + 1e-5f);
    const float xe[4] = { xv.x, xv.y, xv.z, xv.w };
    #pragma unroll
    for (int e = 0; e < 4; e++) {
        const int c = tid * 4 + e;
        out[(size_t)row * HDIM + c] = (_Float16)(((xe[e] - mean) * inv) * g[c] + b[c]);
    }
}

// ---------------------------------------------------------------- Rotary
// Reference quirk: sinu stack puts cos at index 0 but names it `sin`, so the
// effective transform is out[j] = r*sin(rad) + (j even ? -r : r)*cos(rad).
__global__ void rotary_k(_Float16* __restrict__ qk, float scl, int total) {
    const int idx = blockIdx.x * blockDim.x + threadIdx.x;
    if (idx >= total) return;
    const int d    = idx & 63;                 // dim within head
    const int srow = (idx >> 10) & (SEQ - 1);  // sequence position (H==1024)
    float v = (float)qk[idx];
    if (d < 32) {
        const int fi = d & 15;
        const float freq = exp2f((float)fi * (2.3219280948873623f / 15.0f)); // 2^(i*log2(5)/15)
        const float rad  = (float)(srow + 1) * freq * 3.14159265358979f;
        float sn, cs;
        __sincosf(rad, &sn, &cs);
        const float f = (d & 1) ? (sn + cs) : (sn - cs);
        v *= f;
    }
    qk[idx] = (_Float16)(v * scl);
}

// ---------------------------------------------------------------- GEMM
// C[M,N] = act(A[M,K] @ W[K,N] + bias) (+ residual). Block 128x256, wave 64x64.
// ACT: 0 = none, 1 = x*sigmoid(1.702x)
template<int ACT>
__global__ __launch_bounds__(256) void gemm_wmma(
    const _Float16* __restrict__ A, const _Float16* __restrict__ W,
    const float* __restrict__ bias, const float* __restrict__ resid,
    _Float16* __restrict__ outH, float* __restrict__ outF,
    int M, int N, int K)
{
    __shared__ _Float16 As[128][48];   // row-major A tile, padded
    __shared__ _Float16 Bt[256][48];   // B tile stored transposed: Bt[n][k]
    const int tid  = threadIdx.x;
    const int lane = tid & 31;
    const int wave = tid >> 5;
    const int blockRow = blockIdx.y * 128;
    const int blockCol = blockIdx.x * 256;
    const int wrow = (wave >> 2) * 64;
    const int wcol = (wave & 3) * 64;
    const int m16 = lane & 15;
    const int hh  = lane >> 4;

    // per-thread tile-load coordinates
    const int ar  = tid >> 1;            // A tile row
    const int ac  = (tid & 1) * 16;      // A tile col half
    const int kr2 = (tid >> 4) * 2;      // B tile k-row pair: 0,2,..,30
    const int nb  = (tid & 15) * 16;     // B tile col base: 0..240

    v8f acc[4][4];
    const v8f vzero = {};
    #pragma unroll
    for (int i = 0; i < 4; i++)
        #pragma unroll
        for (int j = 0; j < 4; j++) acc[i][j] = vzero;

    for (int kk = 0; kk < K; kk += 32) {
        { // A tile: 128 rows x 32 halves (32B per thread)
            const v8h* src = (const v8h*)(A + (size_t)(blockRow + ar) * K + kk + ac);
            *(v8h*)&As[ar][ac]     = src[0];
            *(v8h*)&As[ar][ac + 8] = src[1];
        }
        { // B tile: 32 k-rows x 256 cols; two k-rows packed per 4B store
            const _Float16* s0 = W + (size_t)(kk + kr2) * N + blockCol + nb;
            const _Float16* s1 = s0 + N;
            v8h a0 = *(const v8h*)(s0);
            v8h a1 = *(const v8h*)(s0 + 8);
            v8h b0 = *(const v8h*)(s1);
            v8h b1 = *(const v8h*)(s1 + 8);
            #pragma unroll
            for (int e = 0; e < 8; e++) {
                v2h p0 = { a0[e], b0[e] };
                v2h p1 = { a1[e], b1[e] };
                *(v2h*)&Bt[nb + e][kr2]     = p0;
                *(v2h*)&Bt[nb + 8 + e][kr2] = p1;
            }
        }
        // prefetch next K-step tiles (weights are L2-resident; hide L2 latency
        // behind the 16 WMMAs of this step) -> global_prefetch_b8
        if (kk + 32 < K) {
            __builtin_prefetch(A + (size_t)(blockRow + ar) * K + kk + 32 + ac, 0, 3);
            __builtin_prefetch(W + (size_t)(kk + 32 + kr2) * N + blockCol + nb, 0, 3);
        }
        __syncthreads();

        v16h af[4], bf[4];
        #pragma unroll
        for (int i = 0; i < 4; i++) { // A frag: lane m = m16, K halves per §7.12.2
            const _Float16* p = &As[wrow + i * 16 + m16][0];
            v8h lo = *(const v8h*)(p + hh * 8);
            v8h hi = *(const v8h*)(p + 16 + hh * 8);
            af[i] = combine8(lo, hi);
        }
        #pragma unroll
        for (int j = 0; j < 4; j++) { // B frag: lane n = m16, K halves per §7.12.2
            const _Float16* p = &Bt[wcol + j * 16 + m16][0];
            v8h lo = *(const v8h*)(p + hh * 8);
            v8h hi = *(const v8h*)(p + 16 + hh * 8);
            bf[j] = combine8(lo, hi);
        }
        #pragma unroll
        for (int i = 0; i < 4; i++)
            #pragma unroll
            for (int j = 0; j < 4; j++)
                acc[i][j] = __builtin_amdgcn_wmma_f32_16x16x32_f16(
                    false, af[i], false, bf[j], (short)0, acc[i][j], false, false);
        __syncthreads();
    }

    // epilogue: bias + activation + residual, f16 or f32 store
    #pragma unroll
    for (int i = 0; i < 4; i++) {
        #pragma unroll
        for (int j = 0; j < 4; j++) {
            const int Cc = blockCol + wcol + j * 16 + m16;
            const float bval = bias ? bias[Cc] : 0.f;
            #pragma unroll
            for (int r = 0; r < 8; r++) {
                const int Rr = blockRow + wrow + i * 16 + hh * 8 + r;
                float vv = acc[i][j][r] + bval;
                if (ACT == 1) vv = vv * (1.f / (1.f + __expf(-1.702f * vv)));
                if (resid) vv += resid[(size_t)Rr * N + Cc];
                if (outF) outF[(size_t)Rr * N + Cc] = vv;
                else      outH[(size_t)Rr * N + Cc] = (_Float16)vv;
            }
        }
    }
}

// ---------------------------------------------------------------- Flash attention
// grid (S/64, NHEAD, BATCH), 128 threads; wave owns 16 q-rows, streams 32-key blocks.
__global__ __launch_bounds__(128) void attn_wmma(
    const _Float16* __restrict__ q, const _Float16* __restrict__ k,
    const _Float16* __restrict__ v, _Float16* __restrict__ att)
{
    __shared__ _Float16 Vt[64][40];        // Vt[dim][key], padded
    __shared__ _Float16 Pst[4][16][48];    // per-wave P staging (16x32 used)
    const int b  = blockIdx.z;
    const int h  = blockIdx.y;
    const int qb = blockIdx.x * 64;
    const int tid  = threadIdx.x;
    const int wave = tid >> 5;
    const int lane = tid & 31;
    const int m16 = lane & 15;
    const int hh  = lane >> 4;

    // loop-invariant Q A-fragments (d-steps 0 and 32)
    v16h aQ[2];
    {
        const int qrow = qb + wave * 16 + m16;
        const _Float16* p = q + ((size_t)(b * SEQ + qrow)) * HDIM + h * DPH;
        #pragma unroll
        for (int ds2 = 0; ds2 < 2; ds2++) {
            v8h lo = *(const v8h*)(p + ds2 * 32 + hh * 8);
            v8h hi = *(const v8h*)(p + ds2 * 32 + 16 + hh * 8);
            aQ[ds2] = combine8(lo, hi);
        }
    }

    float mrun[8], lrun[8];
    #pragma unroll
    for (int r = 0; r < 8; r++) { mrun[r] = -1e30f; lrun[r] = 0.f; }
    const v8f vzero = {};
    v8f Oacc[4];
    #pragma unroll
    for (int dg = 0; dg < 4; dg++) Oacc[dg] = vzero;

    // V transpose-load coordinates: 2 keys packed per 4B store
    const int key2 = ((tid >> 3) & 15) * 2;   // 0,2,..,30
    const int dgb  = (tid & 7) * 8;           // 0..56

    for (int kb = 0; kb < SEQ; kb += 32) {
        { // cooperative V load, transposed into LDS (packed b32 stores)
            const _Float16* p0 = v + ((size_t)(b * SEQ + kb + key2)) * HDIM + h * DPH + dgb;
            const _Float16* p1 = p0 + HDIM;
            v8h a0 = *(const v8h*)p0;
            v8h a1 = *(const v8h*)p1;
            #pragma unroll
            for (int e = 0; e < 8; e++) {
                v2h pr = { a0[e], a1[e] };
                *(v2h*)&Vt[dgb + e][key2] = pr;
            }
        }
        __syncthreads();

        // scores: 16 q-rows x 32 keys, B frags straight from K (contiguous in d)
        v8f sc[2]; sc[0] = vzero; sc[1] = vzero;
        #pragma unroll
        for (int nt = 0; nt < 2; nt++) {
            const int key = kb + nt * 16 + m16;
            const _Float16* p = k + ((size_t)(b * SEQ + key)) * HDIM + h * DPH;
            #pragma unroll
            for (int ds2 = 0; ds2 < 2; ds2++) {
                v8h lo = *(const v8h*)(p + ds2 * 32 + hh * 8);
                v8h hi = *(const v8h*)(p + ds2 * 32 + 16 + hh * 8);
                v16h bK = combine8(lo, hi);
                sc[nt] = __builtin_amdgcn_wmma_f32_16x16x32_f16(
                    false, aQ[ds2], false, bK, (short)0, sc[nt], false, false);
            }
        }

        // online softmax (row spans 16 lanes -> shfl_xor width 16)
        float scale8[8];
        #pragma unroll
        for (int r = 0; r < 8; r++) {
            float mx = fmaxf(sc[0][r], sc[1][r]);
            for (int msk = 1; msk < 16; msk <<= 1) mx = fmaxf(mx, __shfl_xor(mx, msk, 16));
            const float nm = fmaxf(mrun[r], mx);
            scale8[r] = __expf(mrun[r] - nm);
            mrun[r] = nm;
            const float p0 = __expf(sc[0][r] - nm);
            const float p1 = __expf(sc[1][r] - nm);
            sc[0][r] = p0; sc[1][r] = p1;
            float rs = p0 + p1;
            for (int msk = 1; msk < 16; msk <<= 1) rs += __shfl_xor(rs, msk, 16);
            lrun[r] = lrun[r] * scale8[r] + rs;
        }
        #pragma unroll
        for (int dg = 0; dg < 4; dg++)
            #pragma unroll
            for (int r = 0; r < 8; r++) Oacc[dg][r] *= scale8[r];

        // P: C-layout -> A-layout via per-wave LDS round trip
        #pragma unroll
        for (int r = 0; r < 8; r++) {
            Pst[wave][hh * 8 + r][m16]      = (_Float16)sc[0][r];
            Pst[wave][hh * 8 + r][16 + m16] = (_Float16)sc[1][r];
        }
        v16h aP;
        {
            const _Float16* p = &Pst[wave][m16][0];
            v8h lo = *(const v8h*)(p + hh * 8);
            v8h hi = *(const v8h*)(p + 16 + hh * 8);
            aP = combine8(lo, hi);
        }
        #pragma unroll
        for (int dg = 0; dg < 4; dg++) { // P(16x32) @ V(32x64)
            const _Float16* p = &Vt[dg * 16 + m16][0];
            v8h lo = *(const v8h*)(p + hh * 8);
            v8h hi = *(const v8h*)(p + 16 + hh * 8);
            v16h bV = combine8(lo, hi);
            Oacc[dg] = __builtin_amdgcn_wmma_f32_16x16x32_f16(
                false, aP, false, bV, (short)0, Oacc[dg], false, false);
        }
        __syncthreads();  // protect Vt before next key block
    }

    #pragma unroll
    for (int dg = 0; dg < 4; dg++) {
        #pragma unroll
        for (int r = 0; r < 8; r++) {
            const int Rr = qb + wave * 16 + hh * 8 + r;
            const int Cc = h * DPH + dg * 16 + m16;
            att[((size_t)(b * SEQ + Rr)) * HDIM + Cc] = (_Float16)(Oacc[dg][r] / lrun[r]);
        }
    }
}

// ---------------------------------------------------------------- driver
extern "C" void kernel_launch(void* const* d_in, const int* in_sizes, int n_in,
                              void* d_out, int out_size, void* d_ws, size_t ws_size,
                              hipStream_t stream)
{
    (void)in_sizes; (void)n_in; (void)out_size; (void)ws_size;
    const float* x_in = (const float*)d_in[0];
    const float* ln1g = (const float*)d_in[1];
    const float* ln1b = (const float*)d_in[2];
    const float* ln2g = (const float*)d_in[3];
    const float* ln2b = (const float*)d_in[4];
    const float* wq   = (const float*)d_in[5];
    const float* bq   = (const float*)d_in[6];
    const float* wk   = (const float*)d_in[7];
    const float* bk   = (const float*)d_in[8];
    const float* wv   = (const float*)d_in[9];
    const float* bv   = (const float*)d_in[10];
    const float* wo   = (const float*)d_in[11];
    const float* w1   = (const float*)d_in[12];
    const float* b1   = (const float*)d_in[13];
    const float* w2   = (const float*)d_in[14];

    char* ws = (char*)d_ws;
    const size_t MB = 1024u * 1024u;
    float*    xbuf = (float*)(ws);                    // 16 MB f32 residual stream
    _Float16* xl   = (_Float16*)(ws + 16 * MB);       //  8 MB LN output (f16)
    _Float16* attb = (_Float16*)(ws + 24 * MB);       //  8 MB attention output
    _Float16* qbuf = (_Float16*)(ws + 32 * MB);       //  8 MB
    _Float16* kbuf = (_Float16*)(ws + 40 * MB);       //  8 MB
    _Float16* vbuf = (_Float16*)(ws + 48 * MB);       //  8 MB
    _Float16* hid  = (_Float16*)(ws + 32 * MB);       // 32 MB, reuses q/k/v region
    _Float16* wq16 = (_Float16*)(ws + 64 * MB);
    _Float16* wk16 = (_Float16*)(ws + 68 * MB);
    _Float16* wv16 = (_Float16*)(ws + 72 * MB);
    _Float16* wo16 = (_Float16*)(ws + 76 * MB);
    _Float16* w116 = (_Float16*)(ws + 80 * MB);
    _Float16* w216 = (_Float16*)(ws + 96 * MB);       // total footprint ~112 MB

    const int M = BATCH * SEQ;  // 4096

    hipMemcpyAsync(xbuf, x_in, (size_t)M * HDIM * sizeof(float),
                   hipMemcpyDeviceToDevice, stream);

    // one-time weight conversion f32 -> f16 (weights then live in L2: 48 MB < 192 MB)
    const int nHH = LAYERS * HDIM * HDIM;        // 2,097,152
    const int nHE = LAYERS * HDIM * EXPH;        // 8,388,608
    cvt_f32_f16<<<(nHH + 255) / 256, 256, 0, stream>>>(wq, wq16, nHH);
    cvt_f32_f16<<<(nHH + 255) / 256, 256, 0, stream>>>(wk, wk16, nHH);
    cvt_f32_f16<<<(nHH + 255) / 256, 256, 0, stream>>>(wv, wv16, nHH);
    cvt_f32_f16<<<(nHH + 255) / 256, 256, 0, stream>>>(wo, wo16, nHH);
    cvt_f32_f16<<<(nHE + 255) / 256, 256, 0, stream>>>(w1, w116, nHE);
    cvt_f32_f16<<<(nHE + 255) / 256, 256, 0, stream>>>(w2, w216, nHE);

    const dim3 gH(HDIM / 256, M / 128);   // (4, 32)  N=1024 GEMMs
    const dim3 gE(EXPH / 256, M / 128);   // (16, 32) N=4096 GEMM
    const dim3 gA(SEQ / 64, NHEAD, BATCH);
    const int totQK = M * HDIM;

    for (int l = 0; l < LAYERS; l++) {
        const _Float16* lwq = wq16 + (size_t)l * HDIM * HDIM;
        const _Float16* lwk = wk16 + (size_t)l * HDIM * HDIM;
        const _Float16* lwv = wv16 + (size_t)l * HDIM * HDIM;
        const _Float16* lwo = wo16 + (size_t)l * HDIM * HDIM;
        const _Float16* lw1 = w116 + (size_t)l * HDIM * EXPH;
        const _Float16* lw2 = w216 + (size_t)l * EXPH * HDIM;

        layernorm_f16k<<<M, 256, 0, stream>>>(xbuf, ln1g + l * HDIM, ln1b + l * HDIM, xl);
        gemm_wmma<0><<<gH, 256, 0, stream>>>(xl, lwq, bq + l * HDIM, nullptr, qbuf, nullptr, M, HDIM, HDIM);
        gemm_wmma<0><<<gH, 256, 0, stream>>>(xl, lwk, bk + l * HDIM, nullptr, kbuf, nullptr, M, HDIM, HDIM);
        gemm_wmma<0><<<gH, 256, 0, stream>>>(xl, lwv, bv + l * HDIM, nullptr, vbuf, nullptr, M, HDIM, HDIM);
        rotary_k<<<(totQK + 255) / 256, 256, 0, stream>>>(qbuf, 0.125f, totQK);  // folds 1/sqrt(64)
        rotary_k<<<(totQK + 255) / 256, 256, 0, stream>>>(kbuf, 1.0f, totQK);
        attn_wmma<<<gA, 128, 0, stream>>>(qbuf, kbuf, vbuf, attb);
        gemm_wmma<0><<<gH, 256, 0, stream>>>(attb, lwo, nullptr, xbuf, nullptr, xbuf, M, HDIM, HDIM);
        layernorm_f16k<<<M, 256, 0, stream>>>(xbuf, ln2g + l * HDIM, ln2b + l * HDIM, xl);
        gemm_wmma<1><<<gE, 256, 0, stream>>>(xl, lw1, b1 + l * EXPH, nullptr, hid, nullptr, M, EXPH, HDIM);
        gemm_wmma<0><<<gH, 256, 0, stream>>>(hid, lw2, nullptr, xbuf, nullptr, xbuf, M, HDIM, EXPH);
    }

    hipMemcpyAsync(d_out, xbuf, (size_t)M * HDIM * sizeof(float),
                   hipMemcpyDeviceToDevice, stream);
}
